// LSTMGenerator_67542655697789
// MI455X (gfx1250) — matrix operand order repaired and checked by reading, not compiled, not run on hardware
//
#include <hip/hip_runtime.h>
#include <hip/hip_bf16.h>

#define BATCH 4096
#define ND    64
#define H     128
#define G4    512   // 4*H
#define TSEQ  192
#define BN_EPS 1e-5f

typedef __attribute__((ext_vector_type(16))) _Float16 v16h;
typedef __attribute__((ext_vector_type(8)))  _Float16 v8h;
typedef __attribute__((ext_vector_type(8)))  float    v8f;

#define LOG2E 1.4426950408889634f

// Branch-free activations on native v_exp_f32 / v_rcp_f32 (no OCML divergent paths)
__device__ __forceinline__ float fsigm(float x) {
    return __builtin_amdgcn_rcpf(1.0f + __builtin_amdgcn_exp2f(-LOG2E * x));
}
__device__ __forceinline__ float ftanh(float x) {
    // tanh(x) = 2*sigmoid(2x) - 1
    return 2.0f * __builtin_amdgcn_rcpf(1.0f + __builtin_amdgcn_exp2f(-2.0f * LOG2E * x)) - 1.0f;
}

// ---------------- Kernel 1: WcT[k][j] = sum_m w_ih[j,m] * fc_w[m,k]  (folded input weight)
__global__ void prep_wc(const float* __restrict__ w_ih, const float* __restrict__ fc_w,
                        float* __restrict__ WcT) {
    int gid = blockIdx.x * 256 + threadIdx.x;           // G4*ND = 32768
    if (gid >= G4 * ND) return;
    int j = gid & (G4 - 1);
    int k = gid >> 9;
    float acc = 0.f;
    for (int m = 0; m < H; ++m) acc += w_ih[j * H + m] * fc_w[m * ND + k];
    WcT[k * G4 + j] = acc;                              // transposed for coalesced use
}

// ---------------- Kernel 2: bc[j] = w_ih[j,:]·fc_b + b_ih[j] + b_hh[j]
__global__ void prep_bc(const float* __restrict__ w_ih, const float* __restrict__ fc_b,
                        const float* __restrict__ b_ih, const float* __restrict__ b_hh,
                        float* __restrict__ bc) {
    int j = blockIdx.x * 256 + threadIdx.x;
    if (j >= G4) return;
    float acc = b_ih[j] + b_hh[j];
    for (int m = 0; m < H; ++m) acc += w_ih[j * H + m] * fc_b[m];
    bc[j] = acc;
}

// ---------------- Kernel 3: x_proj[b,j] = z[b,:]·WcT[:,j] + bc[j]
__global__ void xproj_k(const float* __restrict__ z, const float* __restrict__ WcT,
                        const float* __restrict__ bc, float* __restrict__ xp) {
    int gid = blockIdx.x * 256 + threadIdx.x;           // BATCH*G4
    int j = gid & (G4 - 1);
    int b = gid >> 9;
    float acc = bc[j];
    for (int k = 0; k < ND; ++k) acc += z[b * ND + k] * WcT[k * G4 + j];
    xp[gid] = acc;
}

// ---------------- Kernel 4: the LSTM recurrence (WMMA)
// Grid: 256 WGs x 256 threads. Each WG owns 16 batch rows; each of the 8 waves
// owns 16 hidden units (all 4 gates for them) => 4 N-tiles x 4 K-steps WMMA/wave/step.
__global__ void __launch_bounds__(256) lstm_k(const float* __restrict__ w_hh,
                                              const float* __restrict__ xproj,
                                              const float* __restrict__ out_w,
                                              const float* __restrict__ out_b,
                                              float* __restrict__ out) {
    __shared__ __align__(16) _Float16 hbuf[16][H];   // h_t as f16, A-source
    __shared__ float red[16][8];                     // per-batch-row partial dots per wave

    const int lane = threadIdx.x & 31;
    const int wave = threadIdx.x >> 5;
    const int b0   = blockIdx.x * 16;
    const int u0   = wave * 16;        // this wave's hidden-unit base
    const int ln   = lane & 15;
    const int lh   = lane >> 4;        // half-wave: 0 or 1

    // ---- B fragments (w_hh^T tiles) resident in VGPRs for all timesteps ----
    // 16-bit B 32x16, contiguous k order used here (same as A): lane l holds column N=l%16, K = (l/16)*16 + [0..15].
    v16h Bf[4][4];
    #pragma unroll
    for (int g = 0; g < 4; ++g) {
        int n = g * H + u0 + ln;                       // gate column (i,f,g,o order)
        #pragma unroll
        for (int ks = 0; ks < 4; ++ks) {
            int kb = ks * 32 + lh * 16;
            const float* src = &w_hh[n * H + kb];      // 16 contiguous f32
            v16h f;
            #pragma unroll
            for (int e = 0; e < 16; ++e) f[e] = (_Float16)src[e];
            Bf[g][ks] = f;
        }
    }

    // ---- x_proj in C/D accumulator layout, loaded once ----
    // C/D 32-bit: VGPR r, lanes 0-15 -> M=r, lanes 16-31 -> M=r+8; N=l%16.
    v8f xp[4];
    #pragma unroll
    for (int g = 0; g < 4; ++g) {
        #pragma unroll
        for (int r = 0; r < 8; ++r) {
            int m = r + 8 * lh;
            xp[g][r] = xproj[(b0 + m) * G4 + g * H + u0 + ln];
        }
    }

    const float ow = out_w[u0 + ln];
    const float ob = out_b[0];

    v8f cst;                                           // cell state, f32
    #pragma unroll
    for (int r = 0; r < 8; ++r) cst[r] = 0.f;

    for (int i = threadIdx.x; i < 16 * H; i += 256)    // h0 = 0
        ((_Float16*)hbuf)[i] = (_Float16)0.f;
    __syncthreads();

    for (int t = 0; t < TSEQ; ++t) {
        // ---- A fragments (h tile 16x128 f16) from LDS ----
        // 16-bit A 16x32, contiguous k order used here: lane l -> M=l%16; K = (l/16)*16 + [0..15] (matches B)
        v16h A[4];
        #pragma unroll
        for (int ks = 0; ks < 4; ++ks) {
            int koff = ks * 32 + lh * 16;
            v8h lo = *(const v8h*)&hbuf[ln][koff];
            v8h hi = *(const v8h*)&hbuf[ln][koff + 8];
            A[ks] = __builtin_shufflevector(lo, hi,
                        0,1,2,3,4,5,6,7,8,9,10,11,12,13,14,15);
        }
        __syncthreads();   // all hbuf/red reads done before overwrite

        // ---- gates = x_proj + h @ w_hh^T : 16 WMMAs, 4 independent gate chains ----
        v8f acc[4];
        #pragma unroll
        for (int g = 0; g < 4; ++g) {
            v8f a = xp[g];
            #pragma unroll
            for (int ks = 0; ks < 4; ++ks)
                a = __builtin_amdgcn_wmma_f32_16x16x32_f16(
                        false, A[ks], false, Bf[g][ks], (short)0, a, false, false);
            acc[g] = a;
        }

        // ---- LSTM cell update + new h into LDS + output partial (branch-free) ----
        float p[8];
        #pragma unroll
        for (int r = 0; r < 8; ++r) {
            float ig = fsigm(acc[0][r]);
            float fg = fsigm(acc[1][r]);
            float gg = ftanh(acc[2][r]);
            float og = fsigm(acc[3][r]);
            float c  = fg * cst[r] + ig * gg;
            cst[r]   = c;
            float h  = og * ftanh(c);
            hbuf[r + 8 * lh][u0 + ln] = (_Float16)h;
            p[r] = h * ow;
        }
        // reduce over the 16 units this wave owns (stay inside 16-lane groups)
        #pragma unroll
        for (int r = 0; r < 8; ++r) {
            float s = p[r];
            s += __shfl_xor(s, 1, 32);
            s += __shfl_xor(s, 2, 32);
            s += __shfl_xor(s, 4, 32);
            s += __shfl_xor(s, 8, 32);
            p[r] = s;
        }
        if (ln == 0) {
            #pragma unroll
            for (int r = 0; r < 8; ++r) red[r + 8 * lh][wave] = p[r];
        }
        __syncthreads();   // h_{t+1} source + partials visible

        if (threadIdx.x < 16) {       // combine 8 waves -> out[b, t] (pre-BN)
            float s = ob;
            #pragma unroll
            for (int w = 0; w < 8; ++w) s += red[threadIdx.x][w];
            out[(b0 + threadIdx.x) * TSEQ + t] = s;
        }
    }
}

// ---------------- Kernel 5: BatchNorm over batch (axis 0) + sigmoid, in place
__global__ void bn_k(float* __restrict__ out, const float* __restrict__ gamma,
                     const float* __restrict__ beta) {
    int t = blockIdx.x;                                 // feature column 0..191
    __shared__ float ssum[256], ssq[256];
    float s = 0.f, q = 0.f;
    for (int b = threadIdx.x; b < BATCH; b += 256) {
        float v = out[b * TSEQ + t];
        s += v; q += v * v;
    }
    ssum[threadIdx.x] = s; ssq[threadIdx.x] = q;
    __syncthreads();
    for (int off = 128; off > 0; off >>= 1) {
        if (threadIdx.x < off) {
            ssum[threadIdx.x] += ssum[threadIdx.x + off];
            ssq[threadIdx.x]  += ssq[threadIdx.x + off];
        }
        __syncthreads();
    }
    float mean  = ssum[0] * (1.0f / BATCH);
    float var   = ssq[0] * (1.0f / BATCH) - mean * mean;
    float scale = gamma[t] * __builtin_amdgcn_rsqf(var + BN_EPS);
    float shift = beta[t] - mean * scale;
    for (int b = threadIdx.x; b < BATCH; b += 256) {
        float v = out[b * TSEQ + t];
        out[b * TSEQ + t] = fsigm(v * scale + shift);
    }
}

extern "C" void kernel_launch(void* const* d_in, const int* in_sizes, int n_in,
                              void* d_out, int out_size, void* d_ws, size_t ws_size,
                              hipStream_t stream) {
    const float* z     = (const float*)d_in[0];
    const float* fc_w  = (const float*)d_in[1];
    const float* fc_b  = (const float*)d_in[2];
    const float* w_ih  = (const float*)d_in[3];
    const float* w_hh  = (const float*)d_in[4];
    const float* b_ih  = (const float*)d_in[5];
    const float* b_hh  = (const float*)d_in[6];
    const float* out_w = (const float*)d_in[7];
    const float* out_b = (const float*)d_in[8];
    const float* gamma = (const float*)d_in[9];
    const float* beta  = (const float*)d_in[10];
    float* out = (float*)d_out;

    float* ws  = (float*)d_ws;
    float* WcT = ws;                    // 64*512
    float* bc  = WcT + ND * G4;         // 512
    float* xp  = bc + G4;               // 4096*512

    prep_wc<<<(G4 * ND + 255) / 256, 256, 0, stream>>>(w_ih, fc_w, WcT);
    prep_bc<<<2, 256, 0, stream>>>(w_ih, fc_b, b_ih, b_hh, bc);
    xproj_k<<<(BATCH * G4) / 256, 256, 0, stream>>>(z, WcT, bc, xp);
    lstm_k<<<BATCH / 16, 256, 0, stream>>>(w_hh, xp, out_w, out_b, out);
    bn_k<<<TSEQ, 256, 0, stream>>>(out, gamma, beta);
}
